// C6TransNet_9320079032918
// MI455X (gfx1250) — matrix operand order repaired and checked by reading, not compiled, not run on hardware
//
#include <hip/hip_runtime.h>
#include <hip/hip_bf16.h>

// ---------------------------------------------------------------------------
// CDNA5 (gfx1250) C6TransNet forward.
// All matmuls via v_wmma_f32_16x16x32_bf16 (wave32). LDS tiles are laid out so
// every WMMA fragment is one contiguous 32B per-lane load (2x ds_load_b128).
// Global->LDS staging uses gfx1250 async copies, double-buffered so transfers
// overlap WMMA compute. Convs run as pure GEMM over a pre-materialized im2col.
// ---------------------------------------------------------------------------

#define USE_ASYNC_COPY 1

typedef __bf16 bf16_t;
typedef __attribute__((ext_vector_type(16))) __bf16 v16bf;
typedef __attribute__((ext_vector_type(8)))  float  v8f;

#define DEV __device__ __forceinline__

static constexpr int BATCH = 64;
static constexpr int WS    = 19;
static constexpr int NTOK  = WS * WS;      // 361
static constexpr int NPAD  = 384;          // padded token count for conv tiles
static constexpr int EMB   = 256;
static constexpr int HEADS = 8;
static constexpr int HD    = 32;
static constexpr int MTOK  = BATCH * NTOK; // 23104 (multiple of 64)

// ---------------- WMMA helpers ----------------

DEV v8f wmma_bf16(v16bf a, v16bf b, v8f c) {
  return __builtin_amdgcn_wmma_f32_16x16x32_bf16(false, a, false, b, (short)0, c,
                                                 false, false);
}

// Unified fragment load: per-lane 16 contiguous bf16 (32B) at
// row (lane&15), half offset (lane>>4)*16 elements.
// A operand tiles are [m][k] row-major (k contiguous), no swizzle either.
// B operand tiles are [n][k] row-major (k contiguous), no swizzle.
DEV v16bf frag_ld(const bf16_t* base, int ld) {
  int lane = threadIdx.x & 31;
  const bf16_t* p = base + (lane & 15) * ld + ((lane >> 4) << 4);
  return *(const v16bf*)p;
}

DEV int kswz(int k) { return (k & 7) | ((k & 8) << 1) | ((k & 16) >> 1); }

// C/D 16x16 f32: lanes 0-15 -> N=lane, M=v; lanes 16-31 -> N=lane-16, M=8+v.
DEV void store_c_f32(float* C, int ld, v8f acc) {
  int lane = threadIdx.x & 31;
  int n = lane & 15, mh = (lane >> 4) * 8;
#pragma unroll
  for (int v = 0; v < 8; ++v) C[(mh + v) * ld + n] = acc[v];
}

// Stage a 64x32 bf16 tile (fully in-bounds) into LDS in 16B chunks.
// 128 threads, 2 chunks each. Async on CDNA5.
DEV void stage_tile64x32(const bf16_t* __restrict__ g, long ldg,
                         bf16_t* lds, int swz) {
  int tid = threadIdx.x;
#pragma unroll
  for (int c = 0; c < 2; ++c) {
    int e = tid + c * 128;            // chunk 0..255
    int r = e >> 2;                   // row 0..63
    int c0 = (e & 3) << 3;            // 0,8,16,24 (elements)
    int p0 = c0;
    const bf16_t* ga = g + (long)r * ldg + c0;
    bf16_t* la = lds + r * 32 + p0;
#if USE_ASYNC_COPY
    unsigned laddr = (unsigned)(uintptr_t)la;
    asm volatile("global_load_async_to_lds_b128 %0, %1, off"
                 :: "v"(laddr), "v"(ga) : "memory");
#else
    *(int4*)la = *(const int4*)ga;
#endif
  }
}

DEV void stage_wait() {
#if USE_ASYNC_COPY
  asm volatile("s_wait_asynccnt 0" ::: "memory");
#endif
}

// ---------------- elementwise / packing kernels ----------------

__global__ void cvt_f32_bf16_kernel(const float* __restrict__ x,
                                    bf16_t* __restrict__ o, long n) {
  long i = (long)blockIdx.x * 256 + threadIdx.x;
  if (i < n) o[i] = (bf16_t)x[i];
}

// convert with K padding: o[r][k<Kp] = k<Ktot ? w[r][k] : 0
__global__ void cvt_pad_kernel(const float* __restrict__ w, bf16_t* __restrict__ o,
                               int rows, int Ktot, int Kp) {
  long i = (long)blockIdx.x * 256 + threadIdx.x;
  if (i >= (long)rows * Kp) return;
  int k = (int)(i % Kp);
  long r = i / Kp;
  o[i] = (k < Ktot) ? (bf16_t)w[r * Ktot + k] : (bf16_t)0.f;
}

__global__ void bnprep_kernel(const float* __restrict__ g, const float* __restrict__ b,
                              const float* __restrict__ m, const float* __restrict__ v,
                              float* __restrict__ scale, float* __restrict__ shift) {
  int c = threadIdx.x;  // 256
  float s = g[c] * rsqrtf(v[c] + 1e-5f);
  scale[c] = s;
  shift[c] = b[c] - m[c] * s;
}

// ---------------- im2col materialization (once per conv) ----------------
// out[b][pos(384)][k(Kp)] = Xin[b][ci][iy*19+ix] with 3x3/dilation window,
// zero padded for pos>=361, k>=Cin*9, or out-of-board taps.

__global__ void im2col_kernel(const bf16_t* __restrict__ Xin, bf16_t* __restrict__ out,
                              int Cin, int Kp, int dil) {
  long i = (long)blockIdx.x * 256 + threadIdx.x;
  long total = (long)BATCH * NPAD * Kp;
  if (i >= total) return;
  int k = (int)(i % Kp);
  long t = i / Kp;
  int pos = (int)(t % NPAD);
  int b = (int)(t / NPAD);
  int Ktot = Cin * 9;
  bf16_t val = (bf16_t)0.f;
  if (pos < NTOK && k < Ktot) {
    int ci = k / 9, r9 = k - ci * 9;
    int ky = r9 / 3, kx = r9 - ky * 3;
    int py = pos / WS, px = pos - py * WS;
    int iy = py + (ky - 1) * dil, ix = px + (kx - 1) * dil;
    if (iy >= 0 && iy < WS && ix >= 0 && ix < WS)
      val = Xin[((long)b * Cin + ci) * NTOK + iy * WS + ix];
  }
  out[i] = val;
}

// ---------------- conv as GEMM over im2col (WMMA, double-buffered) ----------
// out[b][co][pos] = BN(sum_k W[co][k] * imc[b][pos][k]); optional ReLU.

__global__ void conv_gemm_kernel(const bf16_t* __restrict__ Wt,   // [256][Kp]
                                 const bf16_t* __restrict__ imc,  // [B][384][Kp]
                                 const float* __restrict__ scale,
                                 const float* __restrict__ shift,
                                 float* __restrict__ outF,
                                 bf16_t* __restrict__ outB,
                                 int Kp, int do_relu) {
  __shared__ bf16_t As[2][64 * 32];
  __shared__ bf16_t Bs[2][64 * 32];
  const int b  = blockIdx.z;
  const int m0 = blockIdx.y * 64;
  const int n0 = blockIdx.x * 64;
  const int tid = threadIdx.x;       // 128 threads = 4 waves
  const int w = tid >> 5, wm = w >> 1, wn = w & 1;

  const bf16_t* Abase = Wt + (long)m0 * Kp;
  const bf16_t* Bbase = imc + ((long)b * NPAD + n0) * Kp;

  v8f acc[2][2];
#pragma unroll
  for (int i = 0; i < 2; ++i)
#pragma unroll
    for (int j = 0; j < 2; ++j) acc[i][j] = (v8f){0.f,0.f,0.f,0.f,0.f,0.f,0.f,0.f};

  stage_tile64x32(Abase, Kp, As[0], 1);
  stage_tile64x32(Bbase, Kp, Bs[0], 0);
  stage_wait();
  __syncthreads();

  int cur = 0;
  for (int ks = 0; ks < Kp; ks += 32) {
    int nxt = cur ^ 1;
    if (ks + 32 < Kp) {
      stage_tile64x32(Abase + ks + 32, Kp, As[nxt], 1);
      stage_tile64x32(Bbase + ks + 32, Kp, Bs[nxt], 0);
    }
#pragma unroll
    for (int ms = 0; ms < 2; ++ms) {
      v16bf a = frag_ld(&As[cur][(wm * 32 + ms * 16) * 32], 32);
#pragma unroll
      for (int ns = 0; ns < 2; ++ns) {
        v16bf bb = frag_ld(&Bs[cur][(wn * 32 + ns * 16) * 32], 32);
        acc[ms][ns] = wmma_bf16(a, bb, acc[ms][ns]);
      }
    }
    stage_wait();
    __syncthreads();
    cur = nxt;
  }

  int lane = tid & 31;
  int nloc = lane & 15, mh = (lane >> 4) * 8;
#pragma unroll
  for (int ms = 0; ms < 2; ++ms)
#pragma unroll
    for (int ns = 0; ns < 2; ++ns)
#pragma unroll
      for (int v = 0; v < 8; ++v) {
        int co  = m0 + wm * 32 + ms * 16 + mh + v;
        int pos = n0 + wn * 32 + ns * 16 + nloc;
        if (pos < NTOK) {
          float x = acc[ms][ns][v] * scale[co] + shift[co];
          if (do_relu) x = fmaxf(x, 0.f);
          long idx = ((long)b * 256 + co) * NTOK + pos;
          if (outF) outF[idx] = x;
          if (outB) outB[idx] = (bf16_t)x;
        }
      }
}

// ---------------- generic GEMM (WMMA, double-buffered) ----------------------
// out = act(A @ W^T + bias) [+ res].  A:[M][K], W:[N][K] bf16.
// M,N % 64 == 0, K % 32 == 0 (all call sites).  act: 0 none, 1 relu, 2 gelu.

__global__ void gemm_bf16_kernel(const bf16_t* __restrict__ A,
                                 const bf16_t* __restrict__ Wt,
                                 const float* __restrict__ bias,
                                 const float* __restrict__ res,
                                 float* __restrict__ outF,
                                 bf16_t* __restrict__ outB,
                                 int M, int N, int K, int act) {
  __shared__ bf16_t As[2][64 * 32];
  __shared__ bf16_t Bs[2][64 * 32];
  const int m0 = blockIdx.y * 64;
  const int n0 = blockIdx.x * 64;
  const int tid = threadIdx.x;
  const int w = tid >> 5, wm = w >> 1, wn = w & 1;

  const bf16_t* Abase = A  + (long)m0 * K;
  const bf16_t* Bbase = Wt + (long)n0 * K;

  v8f acc[2][2];
#pragma unroll
  for (int i = 0; i < 2; ++i)
#pragma unroll
    for (int j = 0; j < 2; ++j) acc[i][j] = (v8f){0.f,0.f,0.f,0.f,0.f,0.f,0.f,0.f};

  stage_tile64x32(Abase, K, As[0], 1);
  stage_tile64x32(Bbase, K, Bs[0], 0);
  stage_wait();
  __syncthreads();

  int cur = 0;
  for (int ks = 0; ks < K; ks += 32) {
    int nxt = cur ^ 1;
    if (ks + 32 < K) {
      stage_tile64x32(Abase + ks + 32, K, As[nxt], 1);
      stage_tile64x32(Bbase + ks + 32, K, Bs[nxt], 0);
    }
#pragma unroll
    for (int ms = 0; ms < 2; ++ms) {
      v16bf a = frag_ld(&As[cur][(wm * 32 + ms * 16) * 32], 32);
#pragma unroll
      for (int ns = 0; ns < 2; ++ns) {
        v16bf bb = frag_ld(&Bs[cur][(wn * 32 + ns * 16) * 32], 32);
        acc[ms][ns] = wmma_bf16(a, bb, acc[ms][ns]);
      }
    }
    stage_wait();
    __syncthreads();
    cur = nxt;
  }

  int lane = tid & 31;
  int nloc = lane & 15, mh = (lane >> 4) * 8;
#pragma unroll
  for (int ms = 0; ms < 2; ++ms)
#pragma unroll
    for (int ns = 0; ns < 2; ++ns)
#pragma unroll
      for (int v = 0; v < 8; ++v) {
        int row = m0 + wm * 32 + ms * 16 + mh + v;
        int col = n0 + wn * 32 + ns * 16 + nloc;
        float x = acc[ms][ns][v];
        if (bias) x += bias[col];
        if (act == 1) x = fmaxf(x, 0.f);
        else if (act == 2) x = 0.5f * x * (1.f + erff(x * 0.70710678f));
        long idx = (long)row * N + col;
        if (res)  x += res[idx];
        if (outF) outF[idx] = x;
        if (outB) outB[idx] = (bf16_t)x;
      }
}

// ---------------- fused attention (per batch, head, 32-row tile) ------------
// qkv: [B][N][768] f32 (q|k|v head-major).  rpb: [1369][8].  outB: [B][N][256] bf16.

__global__ void attn_kernel(const float* __restrict__ qkv,
                            const float* __restrict__ rpb,
                            bf16_t* __restrict__ outB) {
  extern __shared__ char smem[];
  bf16_t* Aq  = (bf16_t*)smem;                  // 32x32  q tile, [q][d], d contiguous
  bf16_t* Bk  = Aq + 32 * 32;                   // [384 tok][32 d]
  bf16_t* Bv  = Bk + 384 * 32;                  // [32 d][384 tok]
  bf16_t* ScB = Bv + 32 * 384;                  // 32x384 probs, row-major (tok contiguous)
  float*  Sc  = (float*)(ScB + 32 * 384);       // 32x384 scores
  float*  red = Sc + 32 * 384;                  // 128

  const int rt = blockIdx.x, h = blockIdx.y, b = blockIdx.z;
  const int r0 = rt * 32;
  const int tid = threadIdx.x;                  // 128 threads
  const int w = tid >> 5;
  const float qscale = 0.1767766953f;           // 32^-0.5

  for (int e = tid; e < 32 * 8; e += 128) {
    int r = e >> 3, d0 = (e & 7) << 2;
    int q = r0 + r;
    float4 v4 = {0.f, 0.f, 0.f, 0.f};
    if (q < NTOK) v4 = *(const float4*)&qkv[((long)b * NTOK + q) * 768 + h * HD + d0];
    int p0 = d0;
    Aq[r * 32 + p0 + 0] = (bf16_t)(v4.x * qscale);
    Aq[r * 32 + p0 + 1] = (bf16_t)(v4.y * qscale);
    Aq[r * 32 + p0 + 2] = (bf16_t)(v4.z * qscale);
    Aq[r * 32 + p0 + 3] = (bf16_t)(v4.w * qscale);
  }
  for (int e = tid; e < 384 * 8; e += 128) {
    int tok = e >> 3, d0 = (e & 7) << 2;
    float4 v4 = {0.f, 0.f, 0.f, 0.f};
    if (tok < NTOK)
      v4 = *(const float4*)&qkv[((long)b * NTOK + tok) * 768 + 256 + h * HD + d0];
    Bk[tok * 32 + d0 + 0] = (bf16_t)v4.x;
    Bk[tok * 32 + d0 + 1] = (bf16_t)v4.y;
    Bk[tok * 32 + d0 + 2] = (bf16_t)v4.z;
    Bk[tok * 32 + d0 + 3] = (bf16_t)v4.w;
  }
  for (int e = tid; e < 384 * 8; e += 128) {
    int tok = e >> 3, d0 = (e & 7) << 2;
    float4 v4 = {0.f, 0.f, 0.f, 0.f};
    if (tok < NTOK)
      v4 = *(const float4*)&qkv[((long)b * NTOK + tok) * 768 + 512 + h * HD + d0];
    Bv[(d0 + 0) * 384 + tok] = (bf16_t)v4.x;
    Bv[(d0 + 1) * 384 + tok] = (bf16_t)v4.y;
    Bv[(d0 + 2) * 384 + tok] = (bf16_t)v4.z;
    Bv[(d0 + 3) * 384 + tok] = (bf16_t)v4.w;
  }
  __syncthreads();

  // scores = Q @ K^T : each wave owns a 96-wide strip of keys
  {
    v8f acc[2][6];
#pragma unroll
    for (int i = 0; i < 2; ++i)
#pragma unroll
      for (int j = 0; j < 6; ++j) acc[i][j] = (v8f){0.f,0.f,0.f,0.f,0.f,0.f,0.f,0.f};
#pragma unroll
    for (int ms = 0; ms < 2; ++ms) {
      v16bf a = frag_ld(&Aq[(ms * 16) * 32], 32);
#pragma unroll
      for (int nt = 0; nt < 6; ++nt) {
        v16bf bb = frag_ld(&Bk[(w * 96 + nt * 16) * 32], 32);
        acc[ms][nt] = wmma_bf16(a, bb, acc[ms][nt]);
      }
    }
#pragma unroll
    for (int ms = 0; ms < 2; ++ms)
#pragma unroll
      for (int nt = 0; nt < 6; ++nt)
        store_c_f32(&Sc[(ms * 16) * 384 + w * 96 + nt * 16], 384, acc[ms][nt]);
  }
  __syncthreads();

  // rel-pos bias + row softmax; write probs bf16 (row-major)
  {
    int row = tid & 31, part = tid >> 5;
    int q = r0 + row;
    float* Srow = Sc + row * 384;
    int qy = 0, qx = 0;
    if (q < NTOK) { qy = q / WS; qx = q - qy * WS; }
    float mx = -1e30f;
    if (q < NTOK) {
      for (int n = part * 96; n < part * 96 + 96; ++n) {
        float s;
        if (n < NTOK) {
          int ky = n / WS, kx = n - ky * WS;
          int rel = (qy - ky + WS - 1) * (2 * WS - 1) + (qx - kx + WS - 1);
          s = Srow[n] + rpb[rel * HEADS + h];
        } else s = -1e30f;
        Srow[n] = s;
        mx = fmaxf(mx, s);
      }
    }
    red[part * 32 + row] = mx;
    __syncthreads();
    float rowmax = fmaxf(fmaxf(red[row], red[32 + row]),
                         fmaxf(red[64 + row], red[96 + row]));
    float sum = 0.f;
    if (q < NTOK) {
      for (int n = part * 96; n < part * 96 + 96; ++n) {
        float e = (n < NTOK) ? __expf(Srow[n] - rowmax) : 0.f;
        Srow[n] = e;
        sum += e;
      }
    }
    __syncthreads();
    red[part * 32 + row] = sum;
    __syncthreads();
    float rowsum = red[row] + red[32 + row] + red[64 + row] + red[96 + row];
    float inv = (rowsum > 0.f) ? 1.f / rowsum : 0.f;
    bf16_t* Prow = ScB + row * 384;
    for (int n = part * 96; n < part * 96 + 96; ++n) {
      float p = (q < NTOK) ? Srow[n] * inv : 0.f;
      Prow[n] = (bf16_t)p;
    }
  }
  __syncthreads();

  // out = probs @ V : waves 0,1 take the two 16-row tiles
  if (w < 2) {
    v8f acc[2];
    acc[0] = (v8f){0.f,0.f,0.f,0.f,0.f,0.f,0.f,0.f};
    acc[1] = acc[0];
    for (int kt = 0; kt < 12; ++kt) {
      v16bf a = frag_ld(&ScB[(w * 16) * 384 + kt * 32], 384);
#pragma unroll
      for (int ns = 0; ns < 2; ++ns) {
        v16bf bb = frag_ld(&Bv[(ns * 16) * 384 + kt * 32], 384);
        acc[ns] = wmma_bf16(a, bb, acc[ns]);
      }
    }
    int lane = tid & 31;
    int nloc = lane & 15, mh = (lane >> 4) * 8;
#pragma unroll
    for (int ns = 0; ns < 2; ++ns)
#pragma unroll
      for (int v = 0; v < 8; ++v) {
        int q = r0 + w * 16 + mh + v;
        int d = ns * 16 + nloc;
        if (q < NTOK)
          outB[((long)b * NTOK + q) * EMB + h * HD + d] = (bf16_t)acc[ns][v];
      }
  }
}

// ---------------- LayerNorm (wave per row) ----------------

__global__ void ln_kernel(const float* __restrict__ x, const float* __restrict__ g,
                          const float* __restrict__ bsh, bf16_t* __restrict__ out,
                          int rows) {
  int row = blockIdx.x * 8 + (threadIdx.x >> 5);
  int lane = threadIdx.x & 31;
  if (row >= rows) return;
  const float* xr = x + (long)row * EMB;
  float s = 0.f, s2 = 0.f;
  for (int c = lane; c < EMB; c += 32) { float v = xr[c]; s += v; s2 += v * v; }
#pragma unroll
  for (int o = 16; o > 0; o >>= 1) {
    s  += __shfl_xor(s, o, 32);
    s2 += __shfl_xor(s2, o, 32);
  }
  float mu = s * (1.f / EMB);
  float var = s2 * (1.f / EMB) - mu * mu;
  float inv = rsqrtf(var + 1e-5f);
  bf16_t* orow = out + (long)row * EMB;
  for (int c = lane; c < EMB; c += 32)
    orow[c] = (bf16_t)((xr[c] - mu) * inv * g[c] + bsh[c]);
}

// ---------------- SE path ----------------

__global__ void chan_mean_kernel(const float* __restrict__ x, float* __restrict__ y) {
  int b = blockIdx.x, c = threadIdx.x;
  const float* r = x + ((long)b * 256 + c) * NTOK;
  float s = 0.f;
  for (int i = 0; i < NTOK; ++i) s += r[i];
  y[b * 256 + c] = s * (1.f / NTOK);
}

__global__ void se_mlp_kernel(const float* __restrict__ y, const float* __restrict__ w1,
                              const float* __restrict__ w2, float* __restrict__ s) {
  __shared__ float ys[256];
  __shared__ float hs[16];
  int b = blockIdx.x, t = threadIdx.x;
  ys[t] = y[b * 256 + t];
  __syncthreads();
  if (t < 16) {
    float a = 0.f;
    for (int c = 0; c < 256; ++c) a += w1[t * 256 + c] * ys[c];
    hs[t] = fmaxf(a, 0.f);
  }
  __syncthreads();
  float a = 0.f;
  for (int j = 0; j < 16; ++j) a += w2[t * 16 + j] * hs[j];
  s[b * 256 + t] = 1.f / (1.f + __expf(-a));
}

__global__ void se_apply_kernel(const float* __restrict__ c2, const float* __restrict__ s,
                                const float* __restrict__ resin, float* __restrict__ outF,
                                bf16_t* __restrict__ outB) {
  long i = (long)blockIdx.x * 256 + threadIdx.x;
  if (i >= (long)BATCH * 256 * NTOK) return;
  long bc = i / NTOK;
  float v = fmaxf(c2[i] * s[bc] + resin[i], 0.f);
  outF[i] = v;
  outB[i] = (bf16_t)v;
}

// ---------------- layout change, means, heads ----------------

__global__ void to_tokens_kernel(const float* __restrict__ x, float* __restrict__ xt) {
  long i = (long)blockIdx.x * 256 + threadIdx.x;
  if (i >= (long)BATCH * NTOK * EMB) return;
  int e = (int)(i % EMB);
  long t = i / EMB;
  int n = (int)(t % NTOK);
  int b = (int)(t / NTOK);
  xt[i] = x[((long)b * EMB + e) * NTOK + n];
}

__global__ void token_mean_kernel(const float* __restrict__ xt, float* __restrict__ g) {
  int b = blockIdx.x, e = threadIdx.x;
  float s = 0.f;
  for (int n = 0; n < NTOK; ++n) s += xt[((long)b * NTOK + n) * EMB + e];
  g[b * 256 + e] = s * (1.f / NTOK);
}

__global__ void value_head_kernel(const float* __restrict__ g, const float* __restrict__ w1,
                                  const float* __restrict__ b1, const float* __restrict__ w2,
                                  const float* __restrict__ b2, float* __restrict__ out) {
  __shared__ float gs[256];
  __shared__ float hs[256];
  int b = blockIdx.x, t = threadIdx.x;
  gs[t] = g[b * 256 + t];
  __syncthreads();
  float a = b1[t];
  for (int c = 0; c < 256; ++c) a += w1[t * 256 + c] * gs[c];
  hs[t] = fmaxf(a, 0.f);
  __syncthreads();
  if (t == 0) {
    float v = b2[0];
    for (int j = 0; j < 256; ++j) v += w2[j] * hs[j];
    out[b] = tanhf(v);
  }
}

__global__ void token_head_kernel(const float* __restrict__ xt, const float* __restrict__ emb,
                                  const int* __restrict__ midx, const float* __restrict__ w1,
                                  const float* __restrict__ b1, const float* __restrict__ w2,
                                  const float* __restrict__ b2, float* __restrict__ out) {
  __shared__ float xs[256];
  __shared__ float hs[128];
  long t = blockIdx.x;
  int tid = threadIdx.x;   // 128
  int b = (int)(t / NTOK);
  const float* xr = xt + t * EMB;
  const float* er = emb ? emb + (long)midx[b] * EMB : nullptr;
  for (int c = tid; c < 256; c += 128) xs[c] = xr[c] + (er ? er[c] : 0.f);
  __syncthreads();
  float a = b1[tid];
  for (int c = 0; c < 256; ++c) a += w1[tid * 256 + c] * xs[c];
  hs[tid] = fmaxf(a, 0.f);
  __syncthreads();
  if (tid == 0) {
    float p = b2[0];
    for (int j = 0; j < 128; ++j) p += w2[j] * hs[j];
    out[t] = p;
  }
}

// ---------------------------------------------------------------------------
// Host orchestration
// ---------------------------------------------------------------------------

static constexpr size_t ATTN_SMEM =
    (size_t)(32 * 32 + 384 * 32 + 32 * 384 + 32 * 384) * sizeof(bf16_t) +
    (size_t)(32 * 384 + 128) * sizeof(float);

extern "C" void kernel_launch(void* const* d_in, const int* in_sizes, int n_in,
                              void* d_out, int out_size, void* d_ws, size_t ws_size,
                              hipStream_t stream) {
  (void)in_sizes; (void)n_in; (void)out_size; (void)ws_size;

  auto F = [&](int i) { return (const float*)d_in[i]; };
  const float* x_in     = F(0);
  const int*   move_idx = (const int*)d_in[1];
  const int IDX_CONV_IN = 2, IDX_BN_IN = 3;
  const int RES_BASE = 7;    // 5 x 12
  const int BLK_BASE = 67;   // 6 x 12
  const int IDX_H1W1 = 139, IDX_H1B1 = 140, IDX_H1W2 = 141, IDX_H1B2 = 142;
  const int IDX_M1E  = 143;
  const int IDX_H2W1 = 144, IDX_H2B1 = 145, IDX_H2W2 = 146, IDX_H2B2 = 147;
  const int IDX_VW1 = 148, IDX_VB1 = 149, IDX_VW2 = 150, IDX_VB2 = 151;

  const int KP_IN  = 160;    // conv_in K padded (17*9=153 -> 160)
  const int KP_RES = 2304;   // 256*9

  char* wsp = (char*)d_ws;
  auto alloc = [&](size_t bytes) -> void* {
    void* r = (void*)wsp;
    wsp += (bytes + 255) & ~(size_t)255;
    return r;
  };

  bf16_t* w_conv_in = (bf16_t*)alloc((size_t)256 * KP_IN * sizeof(bf16_t));
  bf16_t* w_res[10];
  for (int i = 0; i < 10; ++i)
    w_res[i] = (bf16_t*)alloc((size_t)256 * KP_RES * sizeof(bf16_t));
  float* bnsc = (float*)alloc((size_t)11 * 256 * sizeof(float));
  float* bnsh = (float*)alloc((size_t)11 * 256 * sizeof(float));
  bf16_t *w_qkv[6], *w_proj[6], *w_m1[6], *w_m2[6];
  for (int j = 0; j < 6; ++j) {
    w_qkv[j]  = (bf16_t*)alloc((size_t)768 * 256 * sizeof(bf16_t));
    w_proj[j] = (bf16_t*)alloc((size_t)256 * 256 * sizeof(bf16_t));
    w_m1[j]   = (bf16_t*)alloc((size_t)1024 * 256 * sizeof(bf16_t));
    w_m2[j]   = (bf16_t*)alloc((size_t)256 * 1024 * sizeof(bf16_t));
  }
  const size_t ACT = (size_t)BATCH * 256 * NTOK;
  bf16_t* x_bf  = (bf16_t*)alloc((size_t)BATCH * 17 * NTOK * sizeof(bf16_t));
  bf16_t* imc   = (bf16_t*)alloc((size_t)BATCH * NPAD * KP_RES * sizeof(bf16_t));
  float*  actF0 = (float*)alloc(ACT * sizeof(float));
  float*  actF1 = (float*)alloc(ACT * sizeof(float));
  bf16_t* actB  = (bf16_t*)alloc(ACT * sizeof(bf16_t));
  bf16_t* actB2 = (bf16_t*)alloc(ACT * sizeof(bf16_t));
  float*  xt    = (float*)alloc((size_t)MTOK * EMB * sizeof(float));
  bf16_t* lnB   = (bf16_t*)alloc((size_t)MTOK * EMB * sizeof(bf16_t));
  float*  qkvF  = (float*)alloc((size_t)MTOK * 768 * sizeof(float));
  bf16_t* attnB = (bf16_t*)alloc((size_t)MTOK * EMB * sizeof(bf16_t));
  bf16_t* hidB  = (bf16_t*)alloc((size_t)MTOK * 1024 * sizeof(bf16_t));
  float*  se_y  = (float*)alloc((size_t)BATCH * 256 * sizeof(float));
  float*  se_s  = (float*)alloc((size_t)BATCH * 256 * sizeof(float));
  float*  gfeat = (float*)alloc((size_t)BATCH * 256 * sizeof(float));

  float* outp = (float*)d_out;   // [p1 23104][p2 23104][value 64]

  hipFuncSetAttribute((const void*)attn_kernel,
                      hipFuncAttributeMaxDynamicSharedMemorySize, (int)ATTN_SMEM);

  auto cvt = [&](const float* src, bf16_t* dst, long n) {
    cvt_f32_bf16_kernel<<<dim3((unsigned)((n + 255) / 256)), dim3(256), 0, stream>>>(src, dst, n);
  };
  auto im2col = [&](const bf16_t* src, int Cin, int Kp, int dil) {
    long total = (long)BATCH * NPAD * Kp;
    im2col_kernel<<<dim3((unsigned)((total + 255) / 256)), dim3(256), 0, stream>>>(
        src, imc, Cin, Kp, dil);
  };

  // ---- weight prep ----
  cvt_pad_kernel<<<dim3((256 * KP_IN + 255) / 256), dim3(256), 0, stream>>>(
      F(IDX_CONV_IN), w_conv_in, 256, 153, KP_IN);
  for (int i = 0; i < 5; ++i) {
    int rb = RES_BASE + 12 * i;
    cvt(F(rb + 0), w_res[2 * i + 0], 256L * KP_RES);
    cvt(F(rb + 5), w_res[2 * i + 1], 256L * KP_RES);
    bnprep_kernel<<<1, 256, 0, stream>>>(F(rb + 1), F(rb + 2), F(rb + 3), F(rb + 4),
                                         bnsc + (1 + 2 * i) * 256, bnsh + (1 + 2 * i) * 256);
    bnprep_kernel<<<1, 256, 0, stream>>>(F(rb + 6), F(rb + 7), F(rb + 8), F(rb + 9),
                                         bnsc + (2 + 2 * i) * 256, bnsh + (2 + 2 * i) * 256);
  }
  bnprep_kernel<<<1, 256, 0, stream>>>(F(IDX_BN_IN), F(IDX_BN_IN + 1), F(IDX_BN_IN + 2),
                                       F(IDX_BN_IN + 3), bnsc, bnsh);
  for (int j = 0; j < 6; ++j) {
    int tb = BLK_BASE + 12 * j;
    cvt(F(tb + 2),  w_qkv[j],  768L * 256);
    cvt(F(tb + 3),  w_proj[j], 256L * 256);
    cvt(F(tb + 8),  w_m1[j],   1024L * 256);
    cvt(F(tb + 10), w_m2[j],   256L * 1024);
  }
  cvt(x_in, x_bf, (long)BATCH * 17 * NTOK);

  // ---- conv stem ----
  im2col(x_bf, 17, KP_IN, 1);
  conv_gemm_kernel<<<dim3(6, 4, BATCH), dim3(128), 0, stream>>>(
      w_conv_in, imc, bnsc, bnsh, actF0, actB, KP_IN, 1);

  // ---- residual SE blocks ----
  const int dils[5] = {1, 2, 3, 1, 1};
  const unsigned seBlocks = (unsigned)((ACT + 255) / 256);
  for (int i = 0; i < 5; ++i) {
    int rb = RES_BASE + 12 * i;
    im2col(actB, 256, KP_RES, dils[i]);
    conv_gemm_kernel<<<dim3(6, 4, BATCH), dim3(128), 0, stream>>>(
        w_res[2 * i], imc, bnsc + (1 + 2 * i) * 256, bnsh + (1 + 2 * i) * 256,
        nullptr, actB2, KP_RES, 1);
    im2col(actB2, 256, KP_RES, 1);
    conv_gemm_kernel<<<dim3(6, 4, BATCH), dim3(128), 0, stream>>>(
        w_res[2 * i + 1], imc, bnsc + (2 + 2 * i) * 256, bnsh + (2 + 2 * i) * 256,
        actF1, nullptr, KP_RES, 0);
    chan_mean_kernel<<<dim3(BATCH), dim3(256), 0, stream>>>(actF1, se_y);
    se_mlp_kernel<<<dim3(BATCH), dim3(256), 0, stream>>>(se_y, F(rb + 10), F(rb + 11), se_s);
    se_apply_kernel<<<dim3(seBlocks), dim3(256), 0, stream>>>(actF1, se_s, actF0, actF0, actB);
  }

  to_tokens_kernel<<<dim3(seBlocks), dim3(256), 0, stream>>>(actF0, xt);

  // ---- transformer blocks ----
  for (int j = 0; j < 6; ++j) {
    int tb = BLK_BASE + 12 * j;
    ln_kernel<<<dim3((MTOK + 7) / 8), dim3(256), 0, stream>>>(xt, F(tb + 0), F(tb + 1), lnB, MTOK);
    gemm_bf16_kernel<<<dim3(768 / 64, MTOK / 64), dim3(128), 0, stream>>>(
        lnB, w_qkv[j], nullptr, nullptr, qkvF, nullptr, MTOK, 768, 256, 0);
    attn_kernel<<<dim3(12, HEADS, BATCH), dim3(128), ATTN_SMEM, stream>>>(
        qkvF, F(tb + 5), attnB);
    gemm_bf16_kernel<<<dim3(256 / 64, MTOK / 64), dim3(128), 0, stream>>>(
        attnB, w_proj[j], F(tb + 4), xt, xt, nullptr, MTOK, 256, 256, 0);
    ln_kernel<<<dim3((MTOK + 7) / 8), dim3(256), 0, stream>>>(xt, F(tb + 6), F(tb + 7), lnB, MTOK);
    gemm_bf16_kernel<<<dim3(1024 / 64, MTOK / 64), dim3(128), 0, stream>>>(
        lnB, w_m1[j], F(tb + 9), nullptr, nullptr, hidB, MTOK, 1024, 256, 2);
    gemm_bf16_kernel<<<dim3(256 / 64, MTOK / 64), dim3(128), 0, stream>>>(
        hidB, w_m2[j], F(tb + 11), xt, xt, nullptr, MTOK, 256, 1024, 0);
  }

  // ---- heads ----
  token_mean_kernel<<<dim3(BATCH), dim3(256), 0, stream>>>(xt, gfeat);
  value_head_kernel<<<dim3(BATCH), dim3(256), 0, stream>>>(
      gfeat, F(IDX_VW1), F(IDX_VB1), F(IDX_VW2), F(IDX_VB2), outp + 2 * MTOK);
  token_head_kernel<<<dim3(MTOK), dim3(128), 0, stream>>>(
      xt, nullptr, move_idx, F(IDX_H1W1), F(IDX_H1B1), F(IDX_H1W2), F(IDX_H1B2), outp);
  token_head_kernel<<<dim3(MTOK), dim3(128), 0, stream>>>(
      xt, F(IDX_M1E), move_idx, F(IDX_H2W1), F(IDX_H2B1), F(IDX_H2W2), F(IDX_H2B2),
      outp + MTOK);
}
